// DNCEncoder_68393059221612
// MI455X (gfx1250) — matrix operand order repaired and checked by reading, not compiled, not run on hardware
//
#include <hip/hip_runtime.h>
#include <hip/hip_bf16.h>

// ---------------- sizes ----------------
#define Bsz 32
#define Tt  256
#define Hh  512
#define NN  5
#define RR  2
#define WW  512
#define IFr 2573          // R*W + 3*W + 5*R + 3
#define IFP 2576          // padded to multiple of 16
#define KG  2048          // gates: K = [x|h|r] = 2048, N = 4*H
#define KA  1536          // H + R*W   (Ar = [h|r])
#define EPSf 1e-6f

typedef __attribute__((ext_vector_type(16))) _Float16 v16h;
typedef __attribute__((ext_vector_type(8)))  float    v8f;

union HF16 { v16h v; float4 f4[2]; };

__device__ __forceinline__ float sigm(float x)      { return 1.f / (1.f + __expf(-x)); }
__device__ __forceinline__ float softplusf(float x) { return (x > 20.f) ? x : log1pf(__expf(x)); }

// Load A fragment (ISA 16-bit A 16x32 layout) and do 4 WMMAs against 4 B rows.
__device__ __forceinline__ void gates_kstep(const _Float16* __restrict__ Ab,
                                            const _Float16* __restrict__ B0,
                                            const _Float16* __restrict__ B1,
                                            const _Float16* __restrict__ B2,
                                            const _Float16* __restrict__ B3,
                                            int kb, v8f& ci, v8f& cf, v8f& cg, v8f& co)
{
    HF16 a, b;
    a.f4[0] = *reinterpret_cast<const float4*>(Ab);
    a.f4[1] = *reinterpret_cast<const float4*>(Ab + 16);
    b.f4[0] = *reinterpret_cast<const float4*>(B0 + kb);
    b.f4[1] = *reinterpret_cast<const float4*>(B0 + kb + 16);
    ci = __builtin_amdgcn_wmma_f32_16x16x32_f16(false, a.v, false, b.v, (short)0, ci, false, false);
    b.f4[0] = *reinterpret_cast<const float4*>(B1 + kb);
    b.f4[1] = *reinterpret_cast<const float4*>(B1 + kb + 16);
    cf = __builtin_amdgcn_wmma_f32_16x16x32_f16(false, a.v, false, b.v, (short)0, cf, false, false);
    b.f4[0] = *reinterpret_cast<const float4*>(B2 + kb);
    b.f4[1] = *reinterpret_cast<const float4*>(B2 + kb + 16);
    cg = __builtin_amdgcn_wmma_f32_16x16x32_f16(false, a.v, false, b.v, (short)0, cg, false, false);
    b.f4[0] = *reinterpret_cast<const float4*>(B3 + kb);
    b.f4[1] = *reinterpret_cast<const float4*>(B3 + kb + 16);
    co = __builtin_amdgcn_wmma_f32_16x16x32_f16(false, a.v, false, b.v, (short)0, co, false, false);
}

// ---------------- fused gates GEMM + LSTM pointwise ----------------
// A[m, k] : k<512 -> emb16[m, tstep, k] ; k>=512 -> Ar[m, k-512]  (Ar = [h|r])
// Wg rows = gate outputs (4*H), cols = [x | h | r] (K=2048).
// Each wave: one m-tile (16 batch rows) x one 16-wide hidden slice j, all 4 gates.
__global__ void __launch_bounds__(128)
gates_lstm_kernel(const _Float16* __restrict__ emb16, int tstep,
                  const _Float16* __restrict__ Wg,
                  const float* __restrict__ b_lstm,
                  float* __restrict__ cS, float* __restrict__ hS,
                  _Float16* __restrict__ Ar)
{
    const int wave = blockIdx.x * 4 + (threadIdx.x >> 5);
    const int lane = threadIdx.x & 31;
    const int tn = wave & 31;                 // 32 j-tiles over H=512
    const int tm = wave >> 5;                 // 2 m-tiles over B=32
    const int mrow = tm * 16 + (lane & 15);
    const int ncol = tn * 16 + (lane & 15);   // hidden index j
    const int offA = (lane < 16) ? 0 : 8;
    const int offB = (lane < 16) ? 0 : 8;

    const _Float16* Ax = emb16 + (size_t)mrow * (Tt * Hh) + (size_t)tstep * Hh + offA;
    const _Float16* Ah = Ar + (size_t)mrow * KA + offA;
    const _Float16* B0 = Wg + (size_t)(0 * Hh + ncol) * KG + offB;
    const _Float16* B1 = Wg + (size_t)(1 * Hh + ncol) * KG + offB;
    const _Float16* B2 = Wg + (size_t)(2 * Hh + ncol) * KG + offB;
    const _Float16* B3 = Wg + (size_t)(3 * Hh + ncol) * KG + offB;

    v8f ci = {}, cf = {}, cg = {}, co = {};
    for (int kb = 0; kb < Hh; kb += 32)       // x portion from embeddings
        gates_kstep(Ax + kb, B0, B1, B2, B3, kb, ci, cf, cg, co);
    for (int kb = Hh; kb < KG; kb += 32)      // [h|r] portion from Ar
        gates_kstep(Ah + (kb - Hh), B0, B1, B2, B3, kb, ci, cf, cg, co);

    const int   mbase = tm * 16 + ((lane < 16) ? 0 : 8);
    const float bi = b_lstm[0 * Hh + ncol];
    const float bf = b_lstm[1 * Hh + ncol];
    const float bg = b_lstm[2 * Hh + ncol];
    const float bo = b_lstm[3 * Hh + ncol];
#pragma unroll
    for (int v = 0; v < 8; ++v) {
        const int m = mbase + v;
        float ig = sigm(ci[v] + bi);
        float fg = sigm(cf[v] + bf);
        float gg = tanhf(cg[v] + bg);
        float og = sigm(co[v] + bo);
        float cn = fg * cS[(size_t)m * Hh + ncol] + ig * gg;
        float hn = og * tanhf(cn);
        cS[(size_t)m * Hh + ncol] = cn;
        hS[(size_t)m * Hh + ncol] = hn;
        Ar[(size_t)m * KA + ncol] = (_Float16)hn;   // h slot of [h|r]
    }
}

// ---------------- generic WMMA GEMM: D[M,N] = A[M,K] * B(stored [N,K])^T + bias ----------------
__global__ void __launch_bounds__(128)
gemm_wmma_kernel(const _Float16* __restrict__ A, int lda,
                 const _Float16* __restrict__ Bw, int ldb,
                 const float* __restrict__ bias,
                 float* __restrict__ D, int ldd,
                 int M, int N, int K)
{
    const int wave   = blockIdx.x * 4 + (threadIdx.x >> 5);
    const int lane   = threadIdx.x & 31;
    const int ntiles = N >> 4;
    const int mtiles = M >> 4;
    const int tm = wave / ntiles;
    const int tn = wave % ntiles;
    if (tm >= mtiles) return;                 // wave-uniform exit, EXEC stays full

    const int mrow = tm * 16 + (lane & 15);
    const int ncol = tn * 16 + (lane & 15);
    const int offA = (lane < 16) ? 0 : 8;
    const int offB = (lane < 16) ? 0 : 8;

    const _Float16* Abase = A  + (size_t)mrow * lda + offA;
    const _Float16* Bbase = Bw + (size_t)ncol * ldb + offB;

    v8f acc = {};
    for (int kb = 0; kb < K; kb += 32) {
        HF16 a, bf;
        a.f4[0]  = *reinterpret_cast<const float4*>(Abase + kb);
        a.f4[1]  = *reinterpret_cast<const float4*>(Abase + kb + 16);
        bf.f4[0] = *reinterpret_cast<const float4*>(Bbase + kb);
        bf.f4[1] = *reinterpret_cast<const float4*>(Bbase + kb + 16);
        acc = __builtin_amdgcn_wmma_f32_16x16x32_f16(false, a.v, false, bf.v,
                                                     (short)0, acc, false, false);
    }

    const int   mbase = tm * 16 + ((lane < 16) ? 0 : 8);
    const float bv    = bias ? bias[ncol] : 0.f;
#pragma unroll
    for (int v = 0; v < 8; ++v)
        D[(size_t)(mbase + v) * ldd + ncol] = acc[v] + bv;
}

// ---------------- prep kernels ----------------
__global__ void embed_kernel(const int* __restrict__ source, const float* __restrict__ emb,
                             _Float16* __restrict__ out)
{
    int i = blockIdx.x * blockDim.x + threadIdx.x;       // B*T*H
    if (i >= Bsz * Tt * Hh) return;
    int j  = i & (Hh - 1);
    int bt = i >> 9;                                     // b*T + t
    int tok = source[bt];
    out[i] = (_Float16)emb[(size_t)tok * Hh + j];
}

// Wg cols reordered: [x (w_ih 0:512) | h (w_hh) | r (w_ih 512:1536)]
__global__ void build_wgates(const float* __restrict__ w_ih, const float* __restrict__ w_hh,
                             _Float16* __restrict__ Wg)
{
    int i = blockIdx.x * blockDim.x + threadIdx.x;       // KG*KG
    if (i >= KG * KG) return;
    int n = i >> 11, k = i & (KG - 1);
    float v;
    if (k < Hh)           v = w_ih[(size_t)n * KA + k];            // x
    else if (k < 2 * Hh)  v = w_hh[(size_t)n * Hh + (k - Hh)];     // h
    else                  v = w_ih[(size_t)n * KA + (k - Hh)];     // r (w_ih col 512 + (k-1024))
    Wg[i] = (_Float16)v;
}

__global__ void build_wxi(const float* __restrict__ w_xi, _Float16* __restrict__ W)
{
    int i = blockIdx.x * blockDim.x + threadIdx.x;       // IFP*Hh, W[n*Hh+k] = w_xi[k][n]
    if (i >= IFP * Hh) return;
    int n = i >> 9, k = i & (Hh - 1);
    W[i] = (n < IFr) ? (_Float16)w_xi[(size_t)k * IFr + n] : (_Float16)0.f;
}

__global__ void build_wout(const float* __restrict__ w_out, _Float16* __restrict__ W)
{
    int i = blockIdx.x * blockDim.x + threadIdx.x;       // Hh*KA, W[n*KA+k] = w_out[k][n]
    if (i >= Hh * KA) return;
    int n = i / KA, k = i % KA;
    W[i] = (_Float16)w_out[(size_t)k * Hh + n];
}

__global__ void build_bxi(const float* __restrict__ b_xi, float* __restrict__ bp)
{
    int i = blockIdx.x * blockDim.x + threadIdx.x;
    if (i < IFP) bp[i] = (i < IFr) ? b_xi[i] : 0.f;
}

// ---------------- DNC memory machinery ----------------
__device__ float block_reduce_256(float v, float* s)
{
    int t = threadIdx.x;
    s[t] = v; __syncthreads();
    if (t < 128) s[t] += s[t + 128]; __syncthreads();
    if (t < 64)  s[t] += s[t + 64];  __syncthreads();
    if (t < 32)  s[t] += s[t + 32];  __syncthreads();
    if (t < 16)  s[t] += s[t + 16];  __syncthreads();
    if (t < 8)   s[t] += s[t + 8];   __syncthreads();
    if (t < 4)   s[t] += s[t + 4];   __syncthreads();
    if (t < 2)   s[t] += s[t + 2];   __syncthreads();
    if (t < 1)   s[t] += s[t + 1];   __syncthreads();
    float r = s[0]; __syncthreads();
    return r;
}

// One block per batch element.
__global__ void __launch_bounds__(256)
dnc_step_kernel(const float* __restrict__ xi_all,
                float* __restrict__ Mg, float* __restrict__ ug, float* __restrict__ pg,
                float* __restrict__ Lg, float* __restrict__ wrg, float* __restrict__ wwg,
                _Float16* __restrict__ Ar)
{
    const int b = blockIdx.x;
    const int t = threadIdx.x;
    const float* xi = xi_all + (size_t)b * IFP;
    float* Mb  = Mg  + (size_t)b * NN * WW;
    float* ub  = ug  + b * NN;
    float* pb  = pg  + b * NN;
    float* Lb  = Lg  + b * NN * NN;
    float* wrb = wrg + b * RR * NN;
    float* wwb = wwg + b * NN;

    __shared__ float sred[256];
    __shared__ float s_br[RR], s_fg[RR], s_pi[RR][3];
    __shared__ float s_bw, s_ga, s_gw, s_kwn, s_krn[RR];
    __shared__ float s_cosw[NN], s_wwn[NN], s_unew[NN], s_pnew[NN], s_pold[NN];
    __shared__ float s_Lold[NN * NN], s_Lnew[NN * NN];
    __shared__ float s_wrold[RR][NN], s_wrnew[RR][NN];
    __shared__ float s_fwd[RR][NN], s_bwd[RR][NN], s_cosr[RR][NN];

    // xi layout: kr 0..1023 | br 1024..1025 | kw 1026..1537 | bw 1538 |
    //            e 1539..2050 | v 2051..2562 | fg 2563..2564 | ga 2565 | gw 2566 | pi 2567..2572
    if (t == 0) {
        for (int r = 0; r < RR; ++r) s_br[r] = 1.f + softplusf(xi[RR * WW + r]);
        s_bw = 1.f + softplusf(xi[1538]);
        for (int r = 0; r < RR; ++r) s_fg[r] = sigm(xi[2563 + r]);
        s_ga = sigm(xi[2565]);
        s_gw = sigm(xi[2566]);
        for (int r = 0; r < RR; ++r) {
            float e0 = xi[2567 + r * 3], e1 = xi[2568 + r * 3], e2 = xi[2569 + r * 3];
            float mx = fmaxf(e0, fmaxf(e1, e2));
            float x0 = __expf(e0 - mx), x1 = __expf(e1 - mx), x2 = __expf(e2 - mx);
            float s = x0 + x1 + x2;
            s_pi[r][0] = x0 / s; s_pi[r][1] = x1 / s; s_pi[r][2] = x2 / s;
        }
    }
    if (t < NN)      s_pold[t] = pb[t];
    if (t < NN * NN) s_Lold[t] = Lb[t];
    if (t < RR * NN) s_wrold[t / NN][t % NN] = wrb[t];
    __syncthreads();

    // write-key norm + write cosine vs OLD M
    {
        float acc = 0.f;
        for (int w = t; w < WW; w += 256) { float k = xi[1026 + w]; acc += k * k; }
        float kk = block_reduce_256(acc, sred);
        if (t == 0) s_kwn = rsqrtf(kk + EPSf);
    }
    for (int n = 0; n < NN; ++n) {
        float da = 0.f, dm = 0.f;
        for (int w = t; w < WW; w += 256) {
            float m = Mb[n * WW + w];
            da += xi[1026 + w] * m;
            dm += m * m;
        }
        float dot = block_reduce_256(da, sred);
        float nm  = block_reduce_256(dm, sred);
        if (t == 0) s_cosw[n] = dot * s_kwn * rsqrtf(nm + EPSf);
    }
    __syncthreads();

    // usage, allocation (sorted over N=5), write weighting
    if (t == 0) {
        float un[NN], uu[NN], aA[NN];
        for (int n = 0; n < NN; ++n) {
            float psi = 1.f;
            for (int r = 0; r < RR; ++r) psi *= (1.f - s_fg[r] * s_wrold[r][n]);
            float u0 = ub[n], w0 = wwb[n];
            un[n] = (u0 + w0 - u0 * w0) * psi;
            uu[n] = EPSf + (1.f - EPSf) * un[n];
        }
        int idx[NN] = {0, 1, 2, 3, 4};
        for (int i = 1; i < NN; ++i) {
            int key = idx[i]; float kv = uu[key]; int j = i - 1;
            while (j >= 0 && uu[idx[j]] > kv) { idx[j + 1] = idx[j]; --j; }
            idx[j + 1] = key;
        }
        float cp = 1.f;
        for (int k = 0; k < NN; ++k) { float su = uu[idx[k]]; aA[idx[k]] = (1.f - su) * cp; cp *= su; }
        float mx = -1e30f, sc[NN];
        for (int n = 0; n < NN; ++n) { sc[n] = s_bw * s_cosw[n]; mx = fmaxf(mx, sc[n]); }
        float ss = 0.f;
        for (int n = 0; n < NN; ++n) { sc[n] = __expf(sc[n] - mx); ss += sc[n]; }
        for (int n = 0; n < NN; ++n) {
            float cw = sc[n] / ss;
            s_wwn[n]  = s_gw * (s_ga * aA[n] + (1.f - s_ga) * cw);
            s_unew[n] = un[n];
        }
    }
    __syncthreads();

    // memory write
    for (int i = t; i < NN * WW; i += 256) {
        int n = i / WW, w = i % WW;
        float e = sigm(xi[1539 + w]);
        float v = xi[2051 + w];
        Mb[i] = Mb[i] * (1.f - s_wwn[n] * e) + s_wwn[n] * v;
    }
    __syncthreads();

    // link matrix (uses OLD p), precedence
    if (t < NN * NN) {
        int i = t / NN, j = t % NN;
        float ln = (1.f - s_wwn[i] - s_wwn[j]) * s_Lold[t] + s_wwn[i] * s_pold[j];
        if (i == j) ln = 0.f;
        s_Lnew[t] = ln;
    }
    if (t == 0) {
        float sw = 0.f;
        for (int n = 0; n < NN; ++n) sw += s_wwn[n];
        for (int j = 0; j < NN; ++j) s_pnew[j] = (1.f - sw) * s_pold[j] + s_wwn[j];
    }
    __syncthreads();
    if (t < RR * NN) {                      // fwd/bwd use NEW L, OLD wr
        int r = t / NN, n = t % NN;
        float f = 0.f, bk = 0.f;
        for (int m = 0; m < NN; ++m) {
            f  += s_Lnew[n * NN + m] * s_wrold[r][m];
            bk += s_Lnew[m * NN + n] * s_wrold[r][m];
        }
        s_fwd[r][n] = f; s_bwd[r][n] = bk;
    }
    __syncthreads();

    // read cosines vs NEW M
    for (int r = 0; r < RR; ++r) {
        float acc = 0.f;
        for (int w = t; w < WW; w += 256) { float k = xi[r * WW + w]; acc += k * k; }
        float kk = block_reduce_256(acc, sred);
        if (t == 0) s_krn[r] = rsqrtf(kk + EPSf);
    }
    for (int n = 0; n < NN; ++n) {
        float dm = 0.f, d0 = 0.f, d1 = 0.f;
        for (int w = t; w < WW; w += 256) {
            float m = Mb[n * WW + w];
            dm += m * m;
            d0 += xi[w] * m;
            d1 += xi[WW + w] * m;
        }
        float nm = block_reduce_256(dm, sred);
        float q0 = block_reduce_256(d0, sred);
        float q1 = block_reduce_256(d1, sred);
        if (t == 0) {
            float rn = rsqrtf(nm + EPSf);
            s_cosr[0][n] = q0 * s_krn[0] * rn;
            s_cosr[1][n] = q1 * s_krn[1] * rn;
        }
    }
    __syncthreads();

    if (t < RR) {
        int r = t;
        float mx = -1e30f, sc[NN];
        for (int n = 0; n < NN; ++n) { sc[n] = s_br[r] * s_cosr[r][n]; mx = fmaxf(mx, sc[n]); }
        float ss = 0.f;
        for (int n = 0; n < NN; ++n) { sc[n] = __expf(sc[n] - mx); ss += sc[n]; }
        for (int n = 0; n < NN; ++n) {
            float cr = sc[n] / ss;
            s_wrnew[r][n] = s_pi[r][0] * s_bwd[r][n] + s_pi[r][1] * cr + s_pi[r][2] * s_fwd[r][n];
        }
    }
    __syncthreads();

    // read vectors r = wr @ M (NEW) -> f16 mirror for next matmuls
    for (int i = t; i < RR * WW; i += 256) {
        int r = i / WW, w = i % WW;
        float acc = 0.f;
        for (int n = 0; n < NN; ++n) acc += s_wrnew[r][n] * Mb[n * WW + w];
        Ar[(size_t)b * KA + Hh + i] = (_Float16)acc;   // r slot of [h|r]
    }

    if (t < NN)      { ub[t] = s_unew[t]; pb[t] = s_pnew[t]; wwb[t] = s_wwn[t]; }
    if (t < NN * NN) Lb[t] = s_Lnew[t];
    if (t < RR * NN) wrb[t] = s_wrnew[t / NN][t % NN];
}

__global__ void final_copy_kernel(const float* __restrict__ h, const float* __restrict__ c,
                                  float* __restrict__ out)
{
    int i = blockIdx.x * blockDim.x + threadIdx.x;       // B*H
    if (i >= Bsz * Hh) return;
    out[(size_t)Bsz * Tt * Hh + i]            = h[i];
    out[(size_t)Bsz * Tt * Hh + Bsz * Hh + i] = c[i];
}

// ---------------- host ----------------
extern "C" void kernel_launch(void* const* d_in, const int* in_sizes, int n_in,
                              void* d_out, int out_size, void* d_ws, size_t ws_size,
                              hipStream_t stream)
{
    (void)in_sizes; (void)n_in; (void)out_size; (void)ws_size;
    const int*   source = (const int*)  d_in[0];
    /* d_in[1] = source_lengths: unused by the reference computation */
    const float* emb    = (const float*)d_in[2];
    const float* w_ih   = (const float*)d_in[3];
    const float* w_hh   = (const float*)d_in[4];
    const float* b_lstm = (const float*)d_in[5];
    const float* w_xi   = (const float*)d_in[6];
    const float* b_xi   = (const float*)d_in[7];
    const float* w_out  = (const float*)d_in[8];
    const float* b_out  = (const float*)d_in[9];
    float* out = (float*)d_out;

    char* ws = (char*)d_ws;
    size_t off = 0;
    auto take = [&](size_t bytes) -> char* {
        char* p = ws + off;
        off += (bytes + 255) & ~(size_t)255;
        return p;
    };
    _Float16* emb16  = (_Float16*)take((size_t)Bsz * Tt * Hh * 2);   // 8 MB
    _Float16* Wg16   = (_Float16*)take((size_t)KG * KG * 2);         // 8 MB  [x|h|r] cols
    _Float16* Wxi16  = (_Float16*)take((size_t)IFP * Hh * 2);        // 2.5 MB
    _Float16* Wout16 = (_Float16*)take((size_t)Hh * KA * 2);         // 1.5 MB
    float*    bxi    = (float*)take((size_t)IFP * 4);
    float*    xi     = (float*)take((size_t)Bsz * IFP * 4);
    char* zstart = ws + off;                                         // zero region start
    _Float16* Ar = (_Float16*)take((size_t)Bsz * KA * 2);            // [h|r] f16
    float* hS  = (float*)take((size_t)Bsz * Hh * 4);
    float* cS  = (float*)take((size_t)Bsz * Hh * 4);
    float* MS  = (float*)take((size_t)Bsz * NN * WW * 4);
    float* uS  = (float*)take((size_t)Bsz * NN * 4);
    float* pS  = (float*)take((size_t)Bsz * NN * 4);
    float* LS  = (float*)take((size_t)Bsz * NN * NN * 4);
    float* wrS = (float*)take((size_t)Bsz * RR * NN * 4);
    float* wwS = (float*)take((size_t)Bsz * NN * 4);
    size_t zbytes = (size_t)((ws + off) - zstart);
    hipMemsetAsync(zstart, 0, zbytes, stream);                       // zero all state each call

    // one-time prep (per launch)
    embed_kernel <<<(Bsz * Tt * Hh + 255) / 256, 256, 0, stream>>>(source, emb, emb16);
    build_wgates<<<(KG * KG + 255) / 256,       256, 0, stream>>>(w_ih, w_hh, Wg16);
    build_wxi   <<<(IFP * Hh + 255) / 256,      256, 0, stream>>>(w_xi, Wxi16);
    build_wout  <<<(Hh * KA + 255) / 256,       256, 0, stream>>>(w_out, Wout16);
    build_bxi   <<<(IFP + 255) / 256,           256, 0, stream>>>(b_xi, bxi);

    for (int tstep = 0; tstep < Tt; ++tstep) {
        // fused: gates GEMM (4 accumulators, A split x|h|r) + LSTM pointwise
        gates_lstm_kernel<<<16, 128, 0, stream>>>(emb16, tstep, Wg16, b_lstm, cS, hS, Ar);

        {   // xi[32,2576] = h[32,512] @ w_xi + b_xi   (A = h slot of Ar, lda = KA)
            int tiles = (Bsz / 16) * (IFP / 16);
            gemm_wmma_kernel<<<(tiles + 3) / 4, 128, 0, stream>>>(
                Ar, KA, Wxi16, Hh, bxi, xi, IFP, Bsz, IFP, Hh);
        }
        dnc_step_kernel<<<Bsz, 256, 0, stream>>>(xi, MS, uS, pS, LS, wrS, wwS, Ar);

        {   // outputs[:,t,:] = [h|r][32,1536] @ w_out + b_out, strided into d_out
            int tiles = (Bsz / 16) * (Hh / 16);
            gemm_wmma_kernel<<<(tiles + 3) / 4, 128, 0, stream>>>(
                Ar, KA, Wout16, KA, b_out, out + (size_t)tstep * Hh, Tt * Hh, Bsz, Hh, KA);
        }
    }

    final_copy_kernel<<<(Bsz * Hh + 255) / 256, 256, 0, stream>>>(hS, cS, out);
}
